// HRSNN_55130200211870
// MI455X (gfx1250) — matrix operand order repaired and checked by reading, not compiled, not run on hardware
//
#include <hip/hip_runtime.h>
#include <hip/hip_bf16.h>
#include <math.h>

typedef __attribute__((ext_vector_type(16))) _Float16 v16h;
typedef __attribute__((ext_vector_type(8)))  float    v8f;

#define BATCH 512
#define CIN   64
#define TLEN  100
#define BT    51200      // BATCH*TLEN
#define HID   10
#define NG    40         // 4*HID
#define F20   20

__device__ __forceinline__ float sigm(float x) { return 1.0f / (1.0f + __expf(-x)); }

// ---------------- K0: transpose (B,C,T) f32 -> (B*T, C) f16 ----------------
__global__ void k_cvt_x(const float* __restrict__ x, _Float16* __restrict__ xt) {
    size_t i = (size_t)blockIdx.x * blockDim.x + threadIdx.x;
    if (i >= (size_t)BT * CIN) return;
    size_t row = i >> 6;        // 64 cols
    size_t c   = i & 63;
    size_t b   = row / TLEN;
    size_t t   = row % TLEN;
    xt[i] = (_Float16)x[(b * CIN + c) * TLEN + t];
}

// ---------------- K1: wih (40,64) f32 -> padded (48,64) f16 ----------------
__global__ void k_cvt_w(const float* __restrict__ wih, _Float16* __restrict__ wf) {
    int i = blockIdx.x * blockDim.x + threadIdx.x;   // 48*64 = 3072
    if (i >= 48 * 64) return;
    int r = i >> 6, c = i & 63;
    wf[i] = (r < NG) ? (_Float16)wih[r * 64 + c] : (_Float16)0.0f;
}

// ---------------- K2: WMMA GEMM gate_pre = xt @ wih^T + bih + bhh ----------
// One wave per (16-row M tile, 16-col N tile). K=64 as two k=32 WMMA steps.
__global__ __launch_bounds__(32) void k_gemm_gates(
        const _Float16* __restrict__ xt, const _Float16* __restrict__ wf,
        const float* __restrict__ bih, const float* __restrict__ bhh,
        float* __restrict__ pre) {
    int tile = blockIdx.x;
    int tm = tile / 3, tn = tile % 3;
    int lane = threadIdx.x;
    int ml = lane & 15;        // M row within tile (A) / N col within tile (B,D)
    int kh = lane >> 4;        // lane-half selects K sub-block

    v8f acc = {};
    const _Float16* arow = xt + (size_t)(tm * 16 + ml) * 64;
    const _Float16* brow = wf + (size_t)(tn * 16 + ml) * 64;   // B[k,n] = wih[n,k]

    #pragma unroll
    for (int ko = 0; ko < 64; ko += 32) {
        v16h a{}, b{};
        // A 16x32 f16, contiguous k order used here (not the ISA table order): lanes0-15 hold K0..15, lanes16-31 K16..31
        #pragma unroll
        for (int v = 0; v < 8; ++v) {
            int kbase = kh * 16 + v * 2;
            a[2 * v]     = arow[ko + kbase];
            a[2 * v + 1] = arow[ko + kbase + 1];
        }
        // B 32x16 f16, same contiguous k order as A: lanes0-15 hold K0..15 of col n, lanes16-31 K16..31
        #pragma unroll
        for (int v = 0; v < 8; ++v) {
            b[2 * v]     = brow[ko + kh * 16 + 2 * v];
            b[2 * v + 1] = brow[ko + kh * 16 + 2 * v + 1];
        }
        acc = __builtin_amdgcn_wmma_f32_16x16x32_f16(
                  false, a, false, b, (short)0, acc, false, false);
    }
    int col = tn * 16 + ml;
    if (col < NG) {
        float bias = bih[col] + bhh[col];
        #pragma unroll
        for (int v = 0; v < 8; ++v) {
            int row = tm * 16 + v + kh * 8;    // D layout: VGPR v -> M=v (+8 for hi lanes)
            pre[(size_t)row * NG + col] = acc[v] + bias;
        }
    }
}

// ---------------- K3: single-wave serial scan over 51200 steps -------------
// LSTM recurrence + sc conv + HR1 (3-tau LIF, conv, s4 LIF, fc+skip) + HR2,
// all fused. Weights cached in LDS; cross-lane comm via LDS (single wave,
// DS ops are program-ordered within a wave).
__global__ __launch_bounds__(32) void k_serial(
        const float* __restrict__ pre,
        const float* __restrict__ whh, const float* __restrict__ scw, const float* __restrict__ scb,
        const float* __restrict__ cw1, const float* __restrict__ cb1,
        const float* __restrict__ fw1, const float* __restrict__ fb1,
        const float* __restrict__ sw1, const float* __restrict__ sb1,
        const float* __restrict__ cw2, const float* __restrict__ cb2,
        const float* __restrict__ fw2, const float* __restrict__ fb2,
        const float* __restrict__ sw2, const float* __restrict__ sb2,
        float* __restrict__ out2) {
    __shared__ float s_whh[NG * HID];
    __shared__ float s_scw[F20 * HID], s_scb[F20];
    __shared__ float s_cw1[F20 * F20 * 3], s_cb1[F20];
    __shared__ float s_fw1[F20 * F20], s_fb1[F20];
    __shared__ float s_sw1[F20 * F20], s_sb1[F20];
    __shared__ float s_cw2[F20 * F20 * 3], s_cb2[F20];
    __shared__ float s_fw2[HID * F20], s_fb2[HID];
    __shared__ float s_sw2[HID * F20], s_sb2[HID];
    __shared__ float s_h[HID], s_co[F20], s_o1[F20], s_s4[F20], s_s4b[F20];
    __shared__ int   s_hl1[F20], s_hl2[F20];

    const int lane = threadIdx.x;
    for (int i = lane; i < NG * HID;      i += 32) s_whh[i] = whh[i];
    for (int i = lane; i < F20 * HID;     i += 32) s_scw[i] = scw[i];
    for (int i = lane; i < F20;           i += 32) { s_scb[i] = scb[i]; s_cb1[i] = cb1[i];
                                                     s_fb1[i] = fb1[i]; s_sb1[i] = sb1[i];
                                                     s_cb2[i] = cb2[i]; }
    for (int i = lane; i < F20 * F20 * 3; i += 32) { s_cw1[i] = cw1[i]; s_cw2[i] = cw2[i]; }
    for (int i = lane; i < F20 * F20;     i += 32) { s_fw1[i] = fw1[i]; s_sw1[i] = sw1[i]; }
    for (int i = lane; i < HID * F20;     i += 32) { s_fw2[i] = fw2[i]; s_sw2[i] = sw2[i]; }
    for (int i = lane; i < HID;           i += 32) { s_fb2[i] = fb2[i]; s_sb2[i] = sb2[i]; }
    if (lane < HID) s_h[lane] = 0.0f;
    __syncthreads();

    const float tau0 = 1.22313016014843f, tau1 = 1.36787944117144f, tau2 = 1.44932896411722f;
    const float th0 = 0.14f, th1 = 0.08f, th2 = 0.06f;
    const float tau4 = 1.36787944117144f, th4 = 0.08f;

    float h = 0.0f, c = 0.0f;                          // LSTM state   (lanes 0..9)
    float v1a = 0.f, v1b = 0.f, v1c = 0.f, v4 = 0.f;   // HR1 LIF      (lanes 0..19)
    float v2a = 0.f, v2b = 0.f, v2c = 0.f, v4b = 0.f;  // HR2 LIF      (lanes 0..19)

    for (int t = 0; t < BT; ++t) {
        const float* pr = pre + (size_t)t * NG;
        __builtin_prefetch(pre + (size_t)(t + 16) * NG, 0, 1);   // global_prefetch_b8

        if (lane < HID) {
            float gi = pr[lane], gf = pr[HID + lane], gg = pr[2 * HID + lane], go = pr[3 * HID + lane];
            #pragma unroll
            for (int k = 0; k < HID; ++k) {
                float hk = s_h[k];
                gi += s_whh[(0 * HID + lane) * HID + k] * hk;
                gf += s_whh[(1 * HID + lane) * HID + k] * hk;
                gg += s_whh[(2 * HID + lane) * HID + k] * hk;
                go += s_whh[(3 * HID + lane) * HID + k] * hk;
            }
            c = sigm(gf) * c + sigm(gi) * tanhf(gg);
            h = sigm(go) * tanhf(c);
        }
        __syncthreads();
        if (lane < HID) s_h[lane] = h;
        __syncthreads();

        if (lane < F20) {
            float acc = s_scb[lane];
            #pragma unroll
            for (int k = 0; k < HID; ++k) acc += s_scw[lane * HID + k] * s_h[k];
            s_co[lane] = acc;
            // HR1: three-tau LIF bank
            v1a += (acc - v1a) / tau0; float sA = (v1a >= th0) ? 1.f : 0.f; v1a *= (1.f - sA);
            v1b += (acc - v1b) / tau1; float sB = (v1b >= th1) ? 1.f : 0.f; v1b *= (1.f - sB);
            v1c += (acc - v1c) / tau2; float sC = (v1c >= th2) ? 1.f : 0.f; v1c *= (1.f - sC);
            s_hl1[lane] = (int)sA | ((int)sB << 1) | ((int)sC << 2);
        }
        __syncthreads();

        if (lane < F20) {                               // HR1 conv + s4 LIF
            float acc = s_cb1[lane];
            #pragma unroll 4
            for (int f = 0; f < F20; ++f) {
                int m = s_hl1[f];
                const float* w = &s_cw1[(lane * F20 + f) * 3];
                if (m & 1) acc += w[0];
                if (m & 2) acc += w[1];
                if (m & 4) acc += w[2];
            }
            v4 += (acc - v4) / tau4; float s4v = (v4 >= th4) ? 1.f : 0.f; v4 *= (1.f - s4v);
            s_s4[lane] = s4v;
        }
        __syncthreads();

        if (lane < F20) {                               // HR1 fc + skip -> out1; HR2 LIF bank
            float acc = s_fb1[lane] + s_sb1[lane];
            #pragma unroll 4
            for (int f = 0; f < F20; ++f)
                acc += s_s4[f] * s_fw1[lane * F20 + f] + s_co[f] * s_sw1[lane * F20 + f];
            s_o1[lane] = acc;
            v2a += (acc - v2a) / tau0; float sA = (v2a >= th0) ? 1.f : 0.f; v2a *= (1.f - sA);
            v2b += (acc - v2b) / tau1; float sB = (v2b >= th1) ? 1.f : 0.f; v2b *= (1.f - sB);
            v2c += (acc - v2c) / tau2; float sC = (v2c >= th2) ? 1.f : 0.f; v2c *= (1.f - sC);
            s_hl2[lane] = (int)sA | ((int)sB << 1) | ((int)sC << 2);
        }
        __syncthreads();

        if (lane < F20) {                               // HR2 conv + s4 LIF
            float acc = s_cb2[lane];
            #pragma unroll 4
            for (int f = 0; f < F20; ++f) {
                int m = s_hl2[f];
                const float* w = &s_cw2[(lane * F20 + f) * 3];
                if (m & 1) acc += w[0];
                if (m & 2) acc += w[1];
                if (m & 4) acc += w[2];
            }
            v4b += (acc - v4b) / tau4; float s4v = (v4b >= th4) ? 1.f : 0.f; v4b *= (1.f - s4v);
            s_s4b[lane] = s4v;
        }
        __syncthreads();

        if (lane < HID) {                               // HR2 fc + skip -> out2[t]
            float acc = s_fb2[lane] + s_sb2[lane];
            #pragma unroll 4
            for (int f = 0; f < F20; ++f)
                acc += s_s4b[f] * s_fw2[lane * F20 + f] + s_o1[f] * s_sw2[lane * F20 + f];
            out2[(size_t)t * HID + lane] = acc;
        }
        __syncthreads();
    }
}

// ---------------- K4: 512-step batch-axis LIF + delta pooling --------------
// thread = (ch, group); owns 5 LIF states (t = g*5 + r). No atomics.
__global__ void k_scan5(const float* __restrict__ out2, float* __restrict__ dp) {
    int tid = blockIdx.x * blockDim.x + threadIdx.x;
    if (tid >= HID * F20) return;     // 200 threads
    int ch = tid / F20, g = tid % F20;
    const float tau4 = 1.36787944117144f, th4 = 0.08f;
    float v[5] = {0.f, 0.f, 0.f, 0.f, 0.f};
    for (int b = 0; b < BATCH; ++b) {
        float s[5];
        #pragma unroll
        for (int r = 0; r < 5; ++r) {
            float x = out2[((size_t)b * TLEN + g * 5 + r) * HID + ch];
            v[r] += (x - v[r]) / tau4;
            float sv = (v[r] >= th4) ? 1.f : 0.f;
            v[r] *= (1.f - sv);
            s[r] = sv;
        }
        dp[(size_t)b * 200 + ch * F20 + g] = 0.5f * (s[3] + s[4] - s[0] - s[1]);
    }
}

// ---------------- K5: per-row avg*dp fuse + partial BN stats ---------------
__global__ void k_fuse(const float* __restrict__ dp, float* __restrict__ fused,
                       float* __restrict__ psum, float* __restrict__ psq) {
    int b = blockIdx.x * blockDim.x + threadIdx.x;
    if (b >= BATCH) return;
    const float* row = dp + (size_t)b * 200;
    float s = 0.f, q = 0.f;
    #pragma unroll 4
    for (int g = 0; g < F20; ++g) {
        float a = 0.f;
        #pragma unroll
        for (int ch = 0; ch < HID; ++ch) a += row[ch * F20 + g];
        a *= 0.1f;
        #pragma unroll
        for (int ch = 0; ch < HID; ++ch) {
            float fv = row[ch * F20 + g] * a;
            fused[(size_t)b * 200 + ch * F20 + g] = fv;
            s += fv; q += fv * fv;
        }
    }
    psum[b] = s; psq[b] = q;
}

// ---------------- K6: deterministic global BN stat reduction ---------------
__global__ void k_stats(const float* __restrict__ psum, const float* __restrict__ psq,
                        float* __restrict__ stats) {
    __shared__ float ss[512], sq[512];
    int tid = threadIdx.x;
    ss[tid] = psum[tid]; sq[tid] = psq[tid];
    __syncthreads();
    for (int ofs = 256; ofs > 0; ofs >>= 1) {
        if (tid < ofs) { ss[tid] += ss[tid + ofs]; sq[tid] += sq[tid + ofs]; }
        __syncthreads();
    }
    if (tid == 0) {
        const float n = (float)(BATCH * 200);
        float mean = ss[0] / n;
        float var  = sq[0] / n - mean * mean;
        stats[0] = mean;
        stats[1] = rsqrtf(var + 1e-5f);
    }
}

// ---------------- K7: BN + classifier + log_softmax ------------------------
__global__ void k_logits(const float* __restrict__ fused, const float* __restrict__ stats,
                         const float* __restrict__ gamma, const float* __restrict__ beta,
                         const float* __restrict__ cw, const float* __restrict__ cb,
                         float* __restrict__ out) {
    int b = blockIdx.x * blockDim.x + threadIdx.x;
    if (b >= BATCH) return;
    float mean = stats[0], inv = stats[1], gm = gamma[0], bt = beta[0];
    const float* row = fused + (size_t)b * 200;
    float l[3];
    #pragma unroll
    for (int j = 0; j < 3; ++j) {
        float a = cb[j];
        for (int i = 0; i < 200; ++i) {
            float bn = (row[i] - mean) * inv * gm + bt;
            a += bn * cw[j * 200 + i];
        }
        l[j] = a;
    }
    float m = fmaxf(l[0], fmaxf(l[1], l[2]));
    float z = __expf(l[0] - m) + __expf(l[1] - m) + __expf(l[2] - m);
    float lse = m + __logf(z);
    out[b * 3 + 0] = l[0] - lse;
    out[b * 3 + 1] = l[1] - lse;
    out[b * 3 + 2] = l[2] - lse;
}

// ---------------------------------------------------------------------------
extern "C" void kernel_launch(void* const* d_in, const int* in_sizes, int n_in,
                              void* d_out, int out_size, void* d_ws, size_t ws_size,
                              hipStream_t stream) {
    const float* x        = (const float*)d_in[0];
    const float* lstm_wih = (const float*)d_in[1];
    const float* lstm_whh = (const float*)d_in[2];
    const float* lstm_bih = (const float*)d_in[3];
    const float* lstm_bhh = (const float*)d_in[4];
    const float* sc_w     = (const float*)d_in[5];   // (20,1,10)
    const float* sc_b     = (const float*)d_in[6];
    const float* h1_cw    = (const float*)d_in[7];
    const float* h1_cb    = (const float*)d_in[8];
    const float* h1_fw    = (const float*)d_in[9];
    const float* h1_fb    = (const float*)d_in[10];
    const float* h1_sw    = (const float*)d_in[11];
    const float* h1_sb    = (const float*)d_in[12];
    const float* h2_cw    = (const float*)d_in[13];
    const float* h2_cb    = (const float*)d_in[14];
    const float* h2_fw    = (const float*)d_in[15];
    const float* h2_fb    = (const float*)d_in[16];
    const float* h2_sw    = (const float*)d_in[17];
    const float* h2_sb    = (const float*)d_in[18];
    const float* bn_gamma = (const float*)d_in[19];
    const float* bn_beta  = (const float*)d_in[20];
    const float* cls_w    = (const float*)d_in[21];
    const float* cls_b    = (const float*)d_in[22];

    // workspace layout (256B aligned chunks)
    char* ws = (char*)d_ws;
    size_t off = 0;
    auto take = [&](size_t bytes) { char* p = ws + off; off = (off + bytes + 255) & ~(size_t)255; return p; };
    _Float16* xt_f16  = (_Float16*)take((size_t)BT * CIN * sizeof(_Float16));
    _Float16* wih_f16 = (_Float16*)take((size_t)48 * 64 * sizeof(_Float16));
    float*    pre     = (float*)take((size_t)BT * NG * sizeof(float));
    float*    out2    = (float*)take((size_t)BT * HID * sizeof(float));
    float*    dp      = (float*)take((size_t)BATCH * 200 * sizeof(float));
    float*    fused   = (float*)take((size_t)BATCH * 200 * sizeof(float));
    float*    psum    = (float*)take((size_t)BATCH * sizeof(float));
    float*    psq     = (float*)take((size_t)BATCH * sizeof(float));
    float*    stats   = (float*)take(16);
    (void)ws_size; (void)in_sizes; (void)n_in; (void)out_size;

    // K0/K1: layout + precision conversion
    k_cvt_x<<<((size_t)BT * CIN + 255) / 256, 256, 0, stream>>>(x, xt_f16);
    k_cvt_w<<<(48 * 64 + 255) / 256, 256, 0, stream>>>(lstm_wih, wih_f16);

    // K2: WMMA input-projection GEMM (one wave per 16x16 tile, 3200 M-tiles x 3 N-tiles)
    k_gemm_gates<<<(BT / 16) * 3, 32, 0, stream>>>(xt_f16, wih_f16, lstm_bih, lstm_bhh, pre);

    // K3: fully fused 51200-step serial recurrence (single wave)
    k_serial<<<1, 32, 0, stream>>>(pre,
                                   lstm_whh, sc_w, sc_b,
                                   h1_cw, h1_cb, h1_fw, h1_fb, h1_sw, h1_sb,
                                   h2_cw, h2_cb, h2_fw, h2_fb, h2_sw, h2_sb,
                                   out2);

    // K4: batch-axis LIF scan + delta pooling (200 independent channels, no atomics)
    k_scan5<<<1, 256, 0, stream>>>(out2, dp);

    // K5/K6: fuse + deterministic BN stats
    k_fuse<<<2, 256, 0, stream>>>(dp, fused, psum, psq);
    k_stats<<<1, 512, 0, stream>>>(psum, psq, stats);

    // K7: classifier + log_softmax
    k_logits<<<2, 256, 0, stream>>>(fused, stats, bn_gamma, bn_beta, cls_w, cls_b, (float*)d_out);
}
